// BatchGRU_35966056137478
// MI455X (gfx1250) — matrix operand order repaired and checked by reading, not compiled, not run on hardware
//
#include <hip/hip_runtime.h>

#define Hdim 1024
#define Gnum 256
#define Lpad 128
#define H3   3072

typedef __attribute__((ext_vector_type(16))) _Float16 v16h;
typedef __attribute__((ext_vector_type(8)))  _Float16 v8h;
typedef __attribute__((ext_vector_type(4)))  _Float16 v4h;
typedef __attribute__((ext_vector_type(8)))  float    v8f;

// ---------------------------------------------------------------------------
// WMMA fragment helpers (contiguous k = 16*(lane/16) + s in both A and B, wave32)
// ---------------------------------------------------------------------------

// 16-bit A matrix 16x32 (MxK), row-major source with row stride `stride`.
// lane 0-15 : M=lane,    K = k0..k0+15
// lane16-31 : M=lane-16, K = k0+16..k0+31
static __device__ __forceinline__ v16h load_a_frag(const _Float16* __restrict__ base,
                                                   size_t stride, int m0, int k0, int lane) {
    const int l  = lane & 15;
    const int hi = lane >> 4;
    const _Float16* p = base + (size_t)(m0 + l) * stride + (size_t)(k0 + hi * 16);
    return *(const v16h*)p;
}

// 16-bit B matrix 32x16 (KxN). B[k][n] = W[n][k], W row-major (3072,1024).
// lane 0-15 : N=lane,    K = k0..k0+15   (contiguous in W row)
// lane16-31 : N=lane-16, K = k0+16..k0+31
static __device__ __forceinline__ v16h load_b_frag(const _Float16* __restrict__ w,
                                                   int n0, int k0, int lane) {
    const int l  = lane & 15;
    const int hi = lane >> 4;
    const _Float16* p = w + (size_t)(n0 + l) * Hdim + (size_t)(k0 + hi * 16);
    return *(const v16h*)p;
}

static __device__ __forceinline__ v8f wmma16(v16h a, v16h b, v8f c) {
    return __builtin_amdgcn_wmma_f32_16x16x32_f16(false, a, false, b, (short)0, c, false, false);
}

static __device__ __forceinline__ float sigm(float x) {
    return 1.0f / (1.0f + __expf(-x));
}
static __device__ __forceinline__ float fast_tanh(float x) {
    float e = __expf(2.0f * x);
    return (e - 1.0f) / (e + 1.0f);
}

// ---------------------------------------------------------------------------
// Prep kernels
// ---------------------------------------------------------------------------

__global__ void k_counts(const int* __restrict__ batch, int* __restrict__ counts, int N) {
    int i = blockIdx.x * blockDim.x + threadIdx.x;
    if (i < N) atomicAdd(&counts[batch[i]], 1);
}

__global__ void k_scan(const int* __restrict__ counts, int* __restrict__ start) {
    __shared__ int s[Gnum];
    int t = threadIdx.x;
    int v = counts[t];
    s[t] = v;
    __syncthreads();
    for (int off = 1; off < Gnum; off <<= 1) {
        int x = (t >= off) ? s[t - off] : 0;
        __syncthreads();
        s[t] += x;
        __syncthreads();
    }
    start[t] = s[t] - v;   // exclusive prefix
}

__global__ void k_wconv(const float* __restrict__ src, _Float16* __restrict__ dst) {
    size_t i = ((size_t)blockIdx.x * blockDim.x + threadIdx.x) * 4;
    float4 v = *(const float4*)(src + i);
    v4h o;
    o[0] = (_Float16)v.x; o[1] = (_Float16)v.y;
    o[2] = (_Float16)v.z; o[3] = (_Float16)v.w;
    *(v4h*)(dst + i) = o;
}

// message = relu(h_atom + bias) -> f16 padded msgs (G, L, H)
__global__ void k_scatter(const float* __restrict__ h_atom, const float* __restrict__ bias,
                          const int* __restrict__ batch, const int* __restrict__ start,
                          _Float16* __restrict__ msgs) {
    int i = blockIdx.x;
    int b = batch[i];
    int p = i - start[b];
    if (p >= Lpad) return;
    const float* src = h_atom + (size_t)i * Hdim;
    _Float16* dst = msgs + ((size_t)b * Lpad + p) * Hdim;
    int j = threadIdx.x * 4;
    float4 v  = *(const float4*)(src + j);
    float4 bb = *(const float4*)(bias + j);
    v4h o;
    o[0] = (_Float16)fmaxf(v.x + bb.x, 0.0f);
    o[1] = (_Float16)fmaxf(v.y + bb.y, 0.0f);
    o[2] = (_Float16)fmaxf(v.z + bb.z, 0.0f);
    o[3] = (_Float16)fmaxf(v.w + bb.w, 0.0f);
    *(v4h*)(dst + j) = o;
}

// h0 = segment_max(h_atom) (0 for empty graphs), duplicated for both dirs
__global__ void k_h0(const float* __restrict__ h_atom, const int* __restrict__ counts,
                     const int* __restrict__ start, float* __restrict__ hf32,
                     _Float16* __restrict__ h16) {
    int g = blockIdx.x;
    int j = blockIdx.y * blockDim.x + threadIdx.x;
    int c  = counts[g];
    int s0 = start[g];
    float m = -INFINITY;
    for (int a = 0; a < c; ++a)
        m = fmaxf(m, h_atom[(size_t)(s0 + a) * Hdim + j]);
    if (c == 0) m = 0.0f;
    size_t o = (size_t)g * Hdim + j;
    size_t d = (size_t)Gnum * Hdim;
    hf32[o]     = m;
    hf32[d + o] = m;
    h16[o]      = (_Float16)m;
    h16[d + o]  = (_Float16)m;
}

// ---------------------------------------------------------------------------
// Fused GRU step: both directions, input-proj + hidden-proj + gates + scatter.
// grid = (Hdim/16 = 64 j-tiles, 2 dirs), block = 256 (8 waves),
// wave w owns M-tiles {2w, 2w+1} of the G=256 graph rows.
// ---------------------------------------------------------------------------
__global__ void __launch_bounds__(256) k_step(
    const _Float16* __restrict__ msgs,    // (G, L, H) f16
    const _Float16* __restrict__ w16,     // [dir][w_ih | w_hh] each (3072,1024) f16
    const _Float16* __restrict__ h16cur,  // [dir][G*H]
    const float*    __restrict__ hf32cur, // [dir][G*H]
    _Float16*       __restrict__ h16nxt,
    float*          __restrict__ hf32nxt,
    const float* __restrict__ b_ih_f, const float* __restrict__ b_hh_f,
    const float* __restrict__ b_ih_b, const float* __restrict__ b_hh_b,
    const int* __restrict__ counts, const int* __restrict__ start,
    float* __restrict__ out, int step)
{
    const int lane = threadIdx.x & 31;
    const int wave = threadIdx.x >> 5;
    const int j0   = blockIdx.x * 16;
    const int dir  = blockIdx.y;
    const int tt   = dir ? (Lpad - 1 - step) : step;

    const size_t WSZ = (size_t)H3 * Hdim;
    const _Float16* wih = w16 + (size_t)dir * 2 * WSZ;
    const _Float16* whh = wih + WSZ;
    const _Float16* hA  = h16cur + (size_t)dir * Gnum * Hdim;   // row stride H
    const _Float16* xA  = msgs + (size_t)tt * Hdim;             // row stride L*H
    const float* bi = dir ? b_ih_b : b_ih_f;
    const float* bh = dir ? b_hh_b : b_hh_f;

    // acc[m][0]=r (x+h), acc[m][1]=z (x+h), acc[m][2]=xn, acc[m][3]=hn
    v8f acc[2][4] = {};

    for (int k0 = 0; k0 < Hdim; k0 += 32) {
        v16h bri = load_b_frag(wih, j0,            k0, lane);
        v16h bzi = load_b_frag(wih, j0 + Hdim,     k0, lane);
        v16h bni = load_b_frag(wih, j0 + 2 * Hdim, k0, lane);
        v16h brh = load_b_frag(whh, j0,            k0, lane);
        v16h bzh = load_b_frag(whh, j0 + Hdim,     k0, lane);
        v16h bnh = load_b_frag(whh, j0 + 2 * Hdim, k0, lane);
#pragma unroll
        for (int m = 0; m < 2; ++m) {
            const int g0 = wave * 32 + m * 16;
            v16h ax = load_a_frag(xA, (size_t)Lpad * Hdim, g0, k0, lane);
            v16h ah = load_a_frag(hA, (size_t)Hdim,        g0, k0, lane);
            acc[m][0] = wmma16(ax, bri, acc[m][0]);
            acc[m][0] = wmma16(ah, brh, acc[m][0]);
            acc[m][1] = wmma16(ax, bzi, acc[m][1]);
            acc[m][1] = wmma16(ah, bzh, acc[m][1]);
            acc[m][2] = wmma16(ax, bni, acc[m][2]);
            acc[m][3] = wmma16(ah, bnh, acc[m][3]);
        }
    }

    // C/D layout: element v -> row m0 + v + (lane>=16 ? 8 : 0), col = j0 + (lane&15)
    const int j  = j0 + (lane & 15);
    const int hi = lane >> 4;
    const float br  = bi[j] + bh[j];
    const float bz  = bi[Hdim + j] + bh[Hdim + j];
    const float bxn = bi[2 * Hdim + j];
    const float bhn = bh[2 * Hdim + j];
    const size_t dofs = (size_t)dir * Gnum * Hdim;

#pragma unroll
    for (int m = 0; m < 2; ++m) {
        const int g0 = wave * 32 + m * 16;
#pragma unroll
        for (int v = 0; v < 8; ++v) {
            const int g = g0 + v + hi * 8;
            float r  = sigm(acc[m][0][v] + br);
            float z  = sigm(acc[m][1][v] + bz);
            float n  = fast_tanh(acc[m][2][v] + bxn + r * (acc[m][3][v] + bhn));
            size_t ho = dofs + (size_t)g * Hdim + j;
            float hp = hf32cur[ho];
            float hn2 = (1.0f - z) * n + z * hp;
            hf32nxt[ho] = hn2;
            h16nxt[ho]  = (_Float16)hn2;
            int c = counts[g];
            if (tt < c) {
                int atom = start[g] + tt;
                out[(size_t)atom * (2 * Hdim) + (size_t)dir * Hdim + j] = hn2;
            }
        }
    }
}

// ---------------------------------------------------------------------------
// Host launcher
// ---------------------------------------------------------------------------
extern "C" void kernel_launch(void* const* d_in, const int* in_sizes, int n_in,
                              void* d_out, int out_size, void* d_ws, size_t ws_size,
                              hipStream_t stream) {
    const float* h_atom = (const float*)d_in[0];
    const int*   batch  = (const int*)  d_in[1];
    const float* bias   = (const float*)d_in[2];
    const float* w_ih_f = (const float*)d_in[3];
    const float* w_hh_f = (const float*)d_in[4];
    const float* b_ih_f = (const float*)d_in[5];
    const float* b_hh_f = (const float*)d_in[6];
    const float* w_ih_b = (const float*)d_in[7];
    const float* w_hh_b = (const float*)d_in[8];
    const float* b_ih_b = (const float*)d_in[9];
    const float* b_hh_b = (const float*)d_in[10];
    const int N = in_sizes[0] / Hdim;
    float* out = (float*)d_out;

    const size_t WSZ = (size_t)H3 * Hdim;
    char* p = (char*)d_ws;
    auto carve = [&](size_t bytes) -> char* {
        char* r = p;
        p += (bytes + 255) & ~(size_t)255;
        return r;
    };
    _Float16* msgs  = (_Float16*)carve((size_t)Gnum * Lpad * Hdim * sizeof(_Float16));
    _Float16* w16   = (_Float16*)carve(4 * WSZ * sizeof(_Float16));
    float*    hf0   = (float*)   carve((size_t)2 * Gnum * Hdim * sizeof(float));
    float*    hf1   = (float*)   carve((size_t)2 * Gnum * Hdim * sizeof(float));
    _Float16* h16_0 = (_Float16*)carve((size_t)2 * Gnum * Hdim * sizeof(_Float16));
    _Float16* h16_1 = (_Float16*)carve((size_t)2 * Gnum * Hdim * sizeof(_Float16));
    int*      cnts  = (int*)     carve(Gnum * sizeof(int));
    int*      strt  = (int*)     carve(Gnum * sizeof(int));

    hipMemsetAsync(cnts, 0, Gnum * sizeof(int), stream);
    hipMemsetAsync(msgs, 0, (size_t)Gnum * Lpad * Hdim * sizeof(_Float16), stream);

    k_counts<<<(N + 255) / 256, 256, 0, stream>>>(batch, cnts, N);
    k_scan<<<1, Gnum, 0, stream>>>(cnts, strt);

    const int wblocks = (int)(WSZ / (256 * 4));
    k_wconv<<<wblocks, 256, 0, stream>>>(w_ih_f, w16 + 0 * WSZ);
    k_wconv<<<wblocks, 256, 0, stream>>>(w_hh_f, w16 + 1 * WSZ);
    k_wconv<<<wblocks, 256, 0, stream>>>(w_ih_b, w16 + 2 * WSZ);
    k_wconv<<<wblocks, 256, 0, stream>>>(w_hh_b, w16 + 3 * WSZ);

    k_scatter<<<N, 256, 0, stream>>>(h_atom, bias, batch, strt, msgs);
    k_h0<<<dim3(Gnum, Hdim / 256), 256, 0, stream>>>(h_atom, cnts, strt, hf0, h16_0);

    float*    fc = hf0;   float*    fn = hf1;
    _Float16* hc = h16_0; _Float16* hn = h16_1;
    for (int s = 0; s < Lpad; ++s) {
        k_step<<<dim3(Hdim / 16, 2), 256, 0, stream>>>(
            msgs, w16, hc, fc, hn, fn,
            b_ih_f, b_hh_f, b_ih_b, b_hh_b,
            cnts, strt, out, s);
        float* tf = fc; fc = fn; fn = tf;
        _Float16* th = hc; hc = hn; hn = th;
    }
}
